// Attention_27376121544790
// MI455X (gfx1250) — matrix operand order repaired and checked by reading, not compiled, not run on hardware
//
#include <hip/hip_runtime.h>
#include <hip/hip_bf16.h>
#include <math.h>

typedef __attribute__((ext_vector_type(16))) _Float16 v16h;
typedef __attribute__((ext_vector_type(8)))  _Float16 v8h;
typedef __attribute__((ext_vector_type(8)))  float    v8f;

#define DIMC   1024
#define HEADS  16
#define HEADD  64
#define BATCH  8
#define NTOK   1024
#define SP     (1024 + 4)   // padded LDS pitch for S rows

// ---------------------------------------------------------------------------
// Fragment loaders for v_wmma_f32_16x16x32_f16, wave32 (A/B share contiguous k order, not AMD's table order)
// A 16x32 f16: lane&15 = row M; (lane>>4)*16 = K sub-base; a[0..15]=K+0..15
// B 32x16 f16 (stored K-contiguous per output column): lane&15 = col N;
//   b[i] = B[col][16*(lane>>4) + i]  -> one contiguous v16h per lane
// C 16x16 f32: c[j] -> row j + 8*(lane>>4), col lane&15
// ---------------------------------------------------------------------------
__device__ __forceinline__ v16h load_a_frag(const _Float16* __restrict__ A, int ld, int lane) {
  const _Float16* p = A + (lane & 15) * ld + ((lane >> 4) << 4);
  v8h lo = *(const v8h*)p;
  v8h hi = *(const v8h*)(p + 8);
  return __builtin_shufflevector(lo, hi, 0,1,2,3,4,5,6,7,8,9,10,11,12,13,14,15);
}

__device__ __forceinline__ v16h load_b_frag(const _Float16* __restrict__ Bm, int ld, int lane) {
  const _Float16* p = Bm + (lane & 15) * ld + ((lane >> 4) << 4);
  return *(const v16h*)p;
}

// ---------------------------------------------------------------------------
// Kernel 1: xp = f16(x + gamma*pos_flat); vT[b][h][d][n] = f16(x[b][n][h*64+d])
// ---------------------------------------------------------------------------
__global__ void prep_kernel(const float* __restrict__ x, const float* __restrict__ pos,
                            const float* __restrict__ gamma_p,
                            _Float16* __restrict__ xp, _Float16* __restrict__ vT) {
  int idx = blockIdx.x * blockDim.x + threadIdx.x;      // 0 .. 8M-1
  float g = *gamma_p;
  int b = idx >> 20;
  int n = (idx >> 10) & 1023;
  int c = idx & 1023;
  float xv = x[idx];
  float pv = pos[(size_t)b * (1024 * 1024) + c * 1024 + n];
  xp[idx] = (_Float16)(xv + g * pv);
  int h = c >> 6, d = c & 63;
  vT[((size_t)(b * HEADS + h) * HEADD + d) * NTOK + n] = (_Float16)xv;
}

__global__ void convw_kernel(const float* __restrict__ W, _Float16* __restrict__ Wh) {
  int idx = blockIdx.x * blockDim.x + threadIdx.x;      // 0 .. 1M-1
  Wh[idx] = (_Float16)W[idx];
}

// ---------------------------------------------------------------------------
// WMMA GEMM: C[m][n] = sum_k A[m][k]*Bw[n][k] + bias[n]   (f16 in, f32 out)
// block = 4 waves, 128x128 tile; wave = 64x64 (16 accum frags)
// ---------------------------------------------------------------------------
__global__ void __launch_bounds__(128)
gemm_f16f32_kernel(const _Float16* __restrict__ A, const _Float16* __restrict__ Bw,
                   const float* __restrict__ bias, float* __restrict__ C,
                   int M, int N, int K) {
  const int lane = threadIdx.x & 31;
  const int wid  = threadIdx.x >> 5;
  const int m0 = blockIdx.x * 128 + (wid >> 1) * 64;
  const int n0 = blockIdx.y * 128 + (wid & 1) * 64;

  v8f acc[4][4] = {};
  for (int k0 = 0; k0 < K; k0 += 32) {
    v16h af[4], bf[4];
#pragma unroll
    for (int mi = 0; mi < 4; ++mi)
      af[mi] = load_a_frag(A + (size_t)(m0 + mi * 16) * K + k0, K, lane);
#pragma unroll
    for (int ni = 0; ni < 4; ++ni)
      bf[ni] = load_b_frag(Bw + (size_t)(n0 + ni * 16) * K + k0, K, lane);
#pragma unroll
    for (int mi = 0; mi < 4; ++mi)
#pragma unroll
      for (int ni = 0; ni < 4; ++ni)
        acc[mi][ni] = __builtin_amdgcn_wmma_f32_16x16x32_f16(
            false, af[mi], false, bf[ni], (short)0, acc[mi][ni], false, false);
  }

  const int hi = lane >> 4, col = lane & 15;
#pragma unroll
  for (int mi = 0; mi < 4; ++mi)
#pragma unroll
    for (int ni = 0; ni < 4; ++ni) {
      int n = n0 + ni * 16 + col;
      float bv = bias[n];
#pragma unroll
      for (int j = 0; j < 8; ++j) {
        int m = m0 + mi * 16 + j + hi * 8;
        C[(size_t)m * N + n] = acc[mi][ni][j] + bv;
      }
    }
}

// ---------------------------------------------------------------------------
// L2 normalize each (b,h,n) 64-vector of qk -> q f16 [B][H][N][64]
// one wave per row
// ---------------------------------------------------------------------------
__global__ void norm_kernel(const float* __restrict__ qk, _Float16* __restrict__ q) {
  int gw   = (blockIdx.x * blockDim.x + threadIdx.x) >> 5;  // 0..131071
  int lane = threadIdx.x & 31;
  int b = gw >> 14;
  int h = (gw >> 10) & 15;
  int n = gw & 1023;
  const float* src = qk + ((size_t)(b * NTOK + n) * DIMC + h * HEADD);
  float e0 = src[lane], e1 = src[lane + 32];
  float s = e0 * e0 + e1 * e1;
#pragma unroll
  for (int off = 16; off; off >>= 1) s += __shfl_xor(s, off, 32);
  float inv = rsqrtf(s + 1e-6f);
  _Float16* dst = q + ((size_t)(b * HEADS + h) * NTOK + n) * HEADD;
  dst[lane]      = (_Float16)(e0 * inv);
  dst[lane + 32] = (_Float16)(e1 * inv);
}

// ---------------------------------------------------------------------------
// Attention: block = (qtile, h, b), 4 waves, 64 query rows.
// S[64][1024] f32 lives in dynamic LDS (263 KB -- CDNA5 320KB/WGP).
// Pass1 S=Q K^T * (8/attn_gamma)  (WMMA), Pass2 softmax stats,
// Pass3 O = P V (WMMA, P built on the fly), epilogue blends with x -> f16.
// ---------------------------------------------------------------------------
__global__ void __launch_bounds__(128)
attn_kernel(const _Float16* __restrict__ q, const _Float16* __restrict__ vT,
            const float* __restrict__ x, const float* __restrict__ ag_p,
            const float* __restrict__ sg0_p, const float* __restrict__ sg1_p,
            _Float16* __restrict__ blended) {
  extern __shared__ float S[];              // [64][SP]
  __shared__ float mrow[64];
  __shared__ float lrow[64];

  const int lane = threadIdx.x & 31;
  const int w    = threadIdx.x >> 5;
  const int qt = blockIdx.x;                // 16 query tiles of 64
  const int h  = blockIdx.y;
  const int b  = blockIdx.z;
  const int hi = lane >> 4, col = lane & 15;
  const float factor = 8.0f / (*ag_p);      // 1/scale/attn_gamma, scale = 64^-0.5

  const _Float16* qbh = q + (size_t)(b * HEADS + h) * NTOK * HEADD;
  const int qrow0 = qt * 64 + w * 16;

  v16h qa0 = load_a_frag(qbh + (size_t)qrow0 * HEADD + 0, HEADD, lane);
  v16h qa1 = load_a_frag(qbh + (size_t)qrow0 * HEADD + 32, HEADD, lane);

  // ---- Pass 1: S = Q K^T ----
  for (int kt = 0; kt < 64; ++kt) {
    v16h kb0 = load_b_frag(qbh + (size_t)(kt * 16) * HEADD + 0, HEADD, lane);
    v16h kb1 = load_b_frag(qbh + (size_t)(kt * 16) * HEADD + 32, HEADD, lane);
    v8f acc = {};
    acc = __builtin_amdgcn_wmma_f32_16x16x32_f16(false, qa0, false, kb0, (short)0, acc, false, false);
    acc = __builtin_amdgcn_wmma_f32_16x16x32_f16(false, qa1, false, kb1, (short)0, acc, false, false);
#pragma unroll
    for (int j = 0; j < 8; ++j)
      S[(w * 16 + j + hi * 8) * SP + kt * 16 + col] = acc[j] * factor;
  }
  __syncthreads();

  // ---- Pass 2: per-row max & sum(exp) ----
  {
    int r = w * 16 + col;
    const float* Sr = S + (size_t)r * SP + hi * 512;
    float mx = -3.0e38f;
    for (int c = 0; c < 512; ++c) mx = fmaxf(mx, Sr[c]);
    mx = fmaxf(mx, __shfl_xor(mx, 16, 32));
    float sum = 0.f;
    for (int c = 0; c < 512; ++c) sum += __expf(Sr[c] - mx);
    sum += __shfl_xor(sum, 16, 32);
    if (hi == 0) { mrow[r] = mx; lrow[r] = 1.0f / sum; }
  }
  __syncthreads();

  // ---- Pass 3: O = softmax(S) @ V ----
  {
    const int r = w * 16 + col;                  // A-frag row for this lane
    const float m  = mrow[r];
    const float il = lrow[r];
    const float* Sr = S + (size_t)r * SP;
    const _Float16* vbh = vT + (size_t)(b * HEADS + h) * HEADD * NTOK;

    v8f oacc[4] = {};
    for (int ks = 0; ks < 32; ++ks) {            // 32 keys per step
      const float* sp = Sr + ks * 32 + hi * 16;
      v16h pa;
#pragma unroll
      for (int i = 0; i < 8; ++i) pa[i]     = (_Float16)(__expf(sp[i]      - m) * il);
#pragma unroll
      for (int i = 0; i < 8; ++i) pa[i + 8] = (_Float16)(__expf(sp[i + 8] - m) * il);
#pragma unroll
      for (int dt = 0; dt < 4; ++dt) {
        v16h vb = *(const v16h*)(vbh + (size_t)(dt * 16 + col) * NTOK + ks * 32 + hi * 16);
        oacc[dt] = __builtin_amdgcn_wmma_f32_16x16x32_f16(
            false, pa, false, vb, (short)0, oacc[dt], false, false);
      }
    }

    // epilogue: blended = g0w*out + g1w*x  (f16, feeds final GEMM)
    float g0 = __expf(*sg0_p), g1 = __expf(*sg1_p);
    float tot = g0 + g1;
    float g0w = g0 / tot, g1w = g1 / tot;
#pragma unroll
    for (int dt = 0; dt < 4; ++dt)
#pragma unroll
      for (int j = 0; j < 8; ++j) {
        int n = qt * 64 + w * 16 + j + hi * 8;
        int c = h * HEADD + dt * 16 + col;
        size_t off = (size_t)(b * NTOK + n) * DIMC + c;
        blended[off] = (_Float16)(g0w * oacc[dt][j] + g1w * x[off]);
      }
  }
}

// ---------------------------------------------------------------------------
// Launch
// inputs: 0:x 1:pos 2:W 3:b 4:gamma 5:attn_gamma 6:sum_gamma0 7:sum_gamma1
// ---------------------------------------------------------------------------
extern "C" void kernel_launch(void* const* d_in, const int* in_sizes, int n_in,
                              void* d_out, int out_size, void* d_ws, size_t ws_size,
                              hipStream_t stream) {
  const float* x   = (const float*)d_in[0];
  const float* pos = (const float*)d_in[1];
  const float* W   = (const float*)d_in[2];
  const float* bv  = (const float*)d_in[3];
  const float* gam = (const float*)d_in[4];
  const float* ag  = (const float*)d_in[5];
  const float* sg0 = (const float*)d_in[6];
  const float* sg1 = (const float*)d_in[7];
  float* out = (float*)d_out;

  char* ws = (char*)d_ws;
  const size_t MB = 1024ull * 1024ull;
  _Float16* xp = (_Float16*)(ws);              // 16 MB  [8192][1024] f16
  _Float16* Wh = (_Float16*)(ws + 16 * MB);    //  2 MB  [1024][1024] f16
  float*    qk = (float*)   (ws + 18 * MB);    // 32 MB  [8192][1024] f32
  _Float16* qh = (_Float16*)(ws + 50 * MB);    // 16 MB  [B][H][N][64] f16
  _Float16* vT = (_Float16*)(ws + 66 * MB);    // 16 MB  [B][H][64][N] f16
  _Float16* bl = (_Float16*)(ws + 82 * MB);    // 16 MB  [8192][1024] f16

  // 1) xp, vT
  prep_kernel<<<(BATCH * NTOK * DIMC) / 256, 256, 0, stream>>>(x, pos, gam, xp, vT);
  // 2) W -> f16
  convw_kernel<<<(DIMC * DIMC) / 256, 256, 0, stream>>>(W, Wh);
  // 3) qk = xp @ W^T + b
  gemm_f16f32_kernel<<<dim3(64, 8), 128, 0, stream>>>(xp, Wh, bv, qk,
                                                      BATCH * NTOK, DIMC, DIMC);
  // 4) l2-normalize heads -> q (== k)
  norm_kernel<<<(BATCH * HEADS * NTOK * 32) / 256, 256, 0, stream>>>(qk, qh);
  // 5) attention + blend -> bl
  attn_kernel<<<dim3(16, HEADS, BATCH), 128, 64 * SP * sizeof(float), stream>>>(
      qh, vT, x, ag, sg0, sg1, bl);
  // 6) final = bl @ W^T + b
  gemm_f16f32_kernel<<<dim3(64, 8), 128, 0, stream>>>(bl, Wh, bv, out,
                                                      BATCH * NTOK, DIMC, DIMC);
}
